// SentenceMatrixLayer_1425929142871
// MI455X (gfx1250) — matrix operand order repaired and checked by reading, not compiled, not run on hardware
//
#include <hip/hip_runtime.h>

typedef __attribute__((ext_vector_type(16))) _Float16 v16h;
typedef __attribute__((ext_vector_type(8)))  float    v8f;
typedef __attribute__((ext_vector_type(4)))  float    f4;   // raw vector (nontemporal-safe)

constexpr int   kB = 8;
constexpr int   kN = 2048;
constexpr int   kD = 512;
constexpr float kP = 0.6f;

// ---------------------------------------------------------------------------
// Phase 1: si/sj GEMV via v_wmma_f32_16x16x32_f16.
// One wave per 16-row tile of x (flattened B*N = 16384 rows).
//   A (16x32 f16): lane l<16 -> row l, K = k0+{0..7, 16..23}
//                  lane l>=16 -> row l-16, K = k0+{8..15, 24..31}
//   B (32x16 f16): lane n (column n): half 0 -> K = k0+{0..7, 16..23}, half 1 -> K = k0+{8..15, 24..31}
//                  column 0 = Wi chunk, column 1 = Wj chunk, other columns 0.
//   D (16x16 f32): VGPR g: lanes 0-15 -> M=g, N=lane; lanes 16-31 -> M=8+g.
// So after 16 accumulations: lane {0,16} column 0 holds si, lane {1,17} holds sj.
// ---------------------------------------------------------------------------
__global__ __launch_bounds__(256)
void gemv_wmma_kernel(const float* __restrict__ x, const float* __restrict__ W,
                      const float* __restrict__ bias,
                      float* __restrict__ s, float* __restrict__ t)
{
  const int lane = threadIdx.x & 31;
  const int wave = threadIdx.x >> 5;
  const int tile = blockIdx.x * 8 + wave;       // 16 rows per tile
  const int row0 = tile << 4;
  const int half = lane >> 4;                   // 0 or 1
  const int l    = lane & 15;
  const int row  = row0 + l;

  const float* __restrict__ xrow = x + (size_t)row * kD;
  const float* __restrict__ wcol = W + ((l == 1) ? kD : 0);  // col0=Wi, col1=Wj
  const float  bmask = (l < 2) ? 1.0f : 0.0f;                // zero unused columns

  v8f c = {};
  for (int k0 = 0; k0 < kD; k0 += 32) {
    v16h a;
    const int ka = k0 + half * 8;
#pragma unroll
    for (int i = 0; i < 8; ++i) a[i]     = (_Float16)xrow[ka + i];
#pragma unroll
    for (int i = 0; i < 8; ++i) a[8 + i] = (_Float16)xrow[ka + 16 + i];

    v16h bm;
    const int kb = k0 + half * 8;
#pragma unroll
    for (int i = 0; i < 8; ++i) bm[i] = (_Float16)(wcol[kb + i] * bmask);
#pragma unroll
    for (int i = 0; i < 8; ++i) bm[8 + i] = (_Float16)(wcol[kb + 16 + i] * bmask);

    // (neg_a, A, neg_b, B, c_mod, C, reuse_a, reuse_b)
    c = __builtin_amdgcn_wmma_f32_16x16x32_f16(false, a, false, bm,
                                               (short)0, c, false, false);
  }

  // Extract columns 0 (si) and 1 (sj); fold bias into s.
  if (l < 2) {
    float* __restrict__ dst = (l == 0) ? s : t;
    const float add  = (l == 0) ? bias[0] : 0.0f;
    const int mbase  = row0 + half * 8;
#pragma unroll
    for (int g = 0; g < 8; ++g) dst[mbase + g] = c[g] + add;
  }
}

// ---------------------------------------------------------------------------
// Phase 2: out[b,i,j] = P*sigmoid(s[b,i]+t[b,j]) + (1-P)*adj[b,i,j]
// Pure HBM streaming: 268 MB at 23.3 TB/s ~ 11.5 us. float4 loads/stores,
// non-temporal on the streaming tensors (they don't fit / shouldn't pollute L2);
// s,t (128 KB) stay cache-resident and are reused N times each.
// ---------------------------------------------------------------------------
__global__ __launch_bounds__(256)
void fuse_kernel(const float* __restrict__ adj,
                 const float* __restrict__ s, const float* __restrict__ t,
                 float* __restrict__ out)
{
  const int total4  = kB * kN * (kN / 4);       // 8,388,608 float4 elements
  int idx = blockIdx.x * 256 + threadIdx.x;
  const int stride = gridDim.x * 256;
  const f4* __restrict__ ap = reinterpret_cast<const f4*>(adj);
  f4* __restrict__ op = reinterpret_cast<f4*>(out);

  for (; idx < total4; idx += stride) {
    const int rowid = idx >> 9;                 // b*N + i   (N/4 = 512)
    const int j4    = (idx & 511) << 2;         // j
    const int bidx  = rowid >> 11;              // b         (N = 2048)

    const float si  = s[rowid];                 // bias already folded in
    const f4 tv = *reinterpret_cast<const f4*>(t + bidx * kN + j4);
    const f4 a4 = __builtin_nontemporal_load(ap + idx);

    f4 r;
    float z;
    z = si + tv.x; r.x = kP * (1.0f / (1.0f + __expf(-z))) + (1.0f - kP) * a4.x;
    z = si + tv.y; r.y = kP * (1.0f / (1.0f + __expf(-z))) + (1.0f - kP) * a4.y;
    z = si + tv.z; r.z = kP * (1.0f / (1.0f + __expf(-z))) + (1.0f - kP) * a4.z;
    z = si + tv.w; r.w = kP * (1.0f / (1.0f + __expf(-z))) + (1.0f - kP) * a4.w;

    __builtin_nontemporal_store(r, op + idx);
  }
}

extern "C" void kernel_launch(void* const* d_in, const int* in_sizes, int n_in,
                              void* d_out, int out_size, void* d_ws, size_t ws_size,
                              hipStream_t stream)
{
  const float* x    = (const float*)d_in[0];   // (8, 2048, 512)
  const float* adj  = (const float*)d_in[1];   // (8, 2048, 2048)
  const float* W    = (const float*)d_in[2];   // (1, 1024)
  const float* bias = (const float*)d_in[3];   // (1,)
  float* out = (float*)d_out;                  // (8, 2048, 2048)

  float* s = (float*)d_ws;                     // kB*kN floats, si + bias
  float* t = s + kB * kN;                      // kB*kN floats, sj

  const int tiles = (kB * kN) / 16;            // 1024 tiles, 8 waves/block
  gemv_wmma_kernel<<<tiles / 8, 256, 0, stream>>>(x, W, bias, s, t);

  const int total4 = kB * kN * (kN / 4);
  int blocks = (total4 + 255) / 256;           // 32768
  if (blocks > 32768) blocks = 32768;
  fuse_kernel<<<blocks, 256, 0, stream>>>(adj, s, t, out);
}
